// GraphiT_GT_LSPE_Layer_87960930222611
// MI455X (gfx1250) — matrix operand order repaired and checked by reading, not compiled, not run on hardware
//
#include <hip/hip_runtime.h>
#include <hip/hip_bf16.h>

#define NNODES 50000
#define NEDGES 800000
#define DIM    64
#define NHEAD  8
#define HDIM   8
#define QS     192   // row stride of fused QKV buffer

typedef __attribute__((ext_vector_type(16))) __bf16 v16bf;
typedef __attribute__((ext_vector_type(8)))  float  v8f;
typedef __attribute__((ext_vector_type(4)))  int    v4i;

// ---- gfx1250 async global->LDS staging (ASYNCcnt path), with safe fallback --
#if defined(__has_builtin)
#if __has_builtin(__builtin_amdgcn_global_load_async_to_lds_b128)
#define HAVE_ASYNC_LDS 1
#endif
#endif

__device__ inline void async_copy_b128(const float* gsrc, float* ldst) {
#ifdef HAVE_ASYNC_LDS
  __builtin_amdgcn_global_load_async_to_lds_b128(
      (__attribute__((address_space(1))) v4i*)gsrc,
      (__attribute__((address_space(3))) v4i*)ldst, 0, 0);
#else
  *(float4*)ldst = *(const float4*)gsrc;
#endif
}

__device__ inline void async_wait0() {
#ifdef HAVE_ASYNC_LDS
#if __has_builtin(__builtin_amdgcn_s_wait_asynccnt)
  __builtin_amdgcn_s_wait_asynccnt(0);
#else
  asm volatile("s_wait_asynccnt 0x0" ::: "memory");
#endif
#endif
}

// ---- WMMA fragment loaders (bf16, 16x16x32) ---------------------------------
// A 16x32 (MxK): lanes 0-15 -> M=lane, K in {k0..k0+7, k0+16..k0+23};
//                lanes 16-31 -> M=lane-16, K offset +8 (ISA 7.12.2 table).
__device__ inline v16bf load_a_frag(const __bf16* As, int ld, int kk, int lane) {
  int m  = lane & 15;
  int lh = lane >> 4;
  const __bf16* row = As + m * ld;
  v16bf a;
#pragma unroll
  for (int j = 0; j < 4; ++j) {
    int k0 = kk + lh * 8 + 2 * j;
    a[2*j]     = row[k0];
    a[2*j + 1] = row[k0 + 1];
    int k1 = k0 + 16;
    a[8 + 2*j]     = row[k1];
    a[8 + 2*j + 1] = row[k1 + 1];
  }
  return a;
}

// Same A layout but sourcing f32 LDS (converted at read time).
__device__ inline v16bf load_a_frag_f32(const float* As, int ld, int kk, int lane) {
  int m  = lane & 15;
  int lh = lane >> 4;
  const float* row = As + m * ld;
  v16bf a;
#pragma unroll
  for (int j = 0; j < 4; ++j) {
    int k0 = kk + lh * 8 + 2 * j;
    a[2*j]     = (__bf16)row[k0];
    a[2*j + 1] = (__bf16)row[k0 + 1];
    int k1 = k0 + 16;
    a[8 + 2*j]     = (__bf16)row[k1];
    a[8 + 2*j + 1] = (__bf16)row[k1 + 1];
  }
  return a;
}

// B 32x16 (KxN): lanes 0-15 -> N=lane, K in {kk..kk+7, kk+16..kk+23}; lanes 16-31 -> K offset +8
// (half-split per ISA sparse-B layout note). WtTile rows are W^T: Wt[n*ld+k]=W[k][n].
__device__ inline v16bf load_b_frag(const __bf16* WtTile, int ld, int kk, int lane) {
  int n  = lane & 15;
  int lh = lane >> 4;
  const __bf16* row = WtTile + n * ld;
  v16bf b;
#pragma unroll
  for (int j = 0; j < 4; ++j) {
    int k = kk + lh * 8 + 2 * j;
    b[2*j]     = row[k];
    b[2*j + 1] = row[k + 1];
    int k1 = k + 16;
    b[8 + 2*j]     = row[k1];
    b[8 + 2*j + 1] = row[k1 + 1];
  }
  return b;
}

// ---- Generic fused GEMM: Y = act(Xn@W + bias) + resid -----------------------
// Xn = X, or X/(z+eps) per (row,head) when USE_ZNORM (folds attention normalize).
// ACT: 0=none, 1=relu, 2=tanh. 256 threads = 8 waves; each wave: 16-row strip.
template<int KDIM, int NCOLS, int ACT, bool USE_BIAS, bool USE_RESID, bool USE_ZNORM>
__global__ void __launch_bounds__(256)
gemm_kernel(const float* __restrict__ X, const float* __restrict__ zvec,
            const float* __restrict__ W,
            const float* __restrict__ bias, const float* __restrict__ resid,
            float* __restrict__ Y, int nrows)
{
  __shared__ __bf16 As[128 * KDIM];
  __shared__ __bf16 WtS[NCOLS * KDIM];
  const int t    = threadIdx.x;
  const int row0 = blockIdx.x * 128;

  for (int i = t; i < KDIM * NCOLS; i += 256) {
    int k = i / NCOLS, n = i % NCOLS;
    WtS[n * KDIM + k] = (__bf16)W[i];
  }
  for (int i = t; i < 128 * KDIM; i += 256) {
    int r = i / KDIM, k = i % KDIM;
    int gr = row0 + r;
    float v = (gr < nrows) ? X[(size_t)gr * KDIM + k] : 0.0f;
    if (USE_ZNORM && gr < nrows)
      v /= (zvec[(size_t)gr * NHEAD + (k >> 3)] + 1e-6f);
    As[i] = (__bf16)v;
  }
  __syncthreads();

  const int wave = t >> 5, lane = t & 31;
  const __bf16* Atile = As + (wave * 16) * KDIM;

  v8f acc[NCOLS / 16] = {};
#pragma unroll
  for (int kk = 0; kk < KDIM; kk += 32) {
    v16bf a = load_a_frag(Atile, KDIM, kk, lane);
#pragma unroll
    for (int nt = 0; nt < NCOLS / 16; ++nt) {
      v16bf b = load_b_frag(WtS + nt * 16 * KDIM, KDIM, kk, lane);
      acc[nt] = __builtin_amdgcn_wmma_f32_16x16x32_bf16(
          false, a, false, b, (short)0, acc[nt], false, false);
    }
  }

  const int lh = lane >> 4, col = lane & 15;
#pragma unroll
  for (int nt = 0; nt < NCOLS / 16; ++nt) {
    int c = nt * 16 + col;
    float bv = USE_BIAS ? bias[c] : 0.0f;
#pragma unroll
    for (int r = 0; r < 8; ++r) {
      int gr = row0 + wave * 16 + lh * 8 + r;
      if (gr < nrows) {
        float v = acc[nt][r] + bv;
        if (ACT == 1) v = fmaxf(v, 0.0f);
        else if (ACT == 2) v = tanhf(v);
        if (USE_RESID) v += resid[(size_t)gr * NCOLS + c];
        Y[(size_t)gr * NCOLS + c] = v;
      }
    }
  }
}

// ---- concat [h|p] -----------------------------------------------------------
__global__ void concat_kernel(const float* __restrict__ h, const float* __restrict__ p,
                              float* __restrict__ hp, int n)
{
  int i = blockIdx.x * blockDim.x + threadIdx.x;
  if (i < n * DIM) {
    int r = i / DIM, c = i % DIM;
    hp[(size_t)r * (2 * DIM) + c]       = h[i];
    hp[(size_t)r * (2 * DIM) + DIM + c] = p[i];
  }
}

// ---- pack [Wq|Wk|Wv] columns into one KDIM x 192 weight ---------------------
__global__ void pack_qkv_kernel(const float* __restrict__ Wq, const float* __restrict__ Wk,
                                const float* __restrict__ Wv, float* __restrict__ Wc, int kdim)
{
  int i = blockIdx.x * blockDim.x + threadIdx.x;
  if (i < kdim * QS) {
    int k = i / QS, c = i % QS;
    const float* srcw = (c < 64) ? Wq : (c < 128) ? Wk : Wv;
    Wc[i] = srcw[k * 64 + (c & 63)];
  }
}

// ---- Fused edge kernel: Ee = e@We (WMMA) -> score -> exp -> scatter ---------
// 64 edges/block, grid.y = channel (0: h-attn, 1: p-attn).
// QKV buffers hold [Q|K|V] per node row (stride QS=192).
__global__ void __launch_bounds__(256)
edge_attn_kernel(const float* __restrict__ e,
                 const int* __restrict__ src, const int* __restrict__ dst,
                 const float* __restrict__ QKVh, const float* __restrict__ QKVp,
                 const float* __restrict__ We_h, const float* __restrict__ We_p,
                 float* __restrict__ wV_h, float* __restrict__ z_h,
                 float* __restrict__ wV_p, float* __restrict__ z_p)
{
  const int chan = blockIdx.y;
  const float* QKV = chan ? QKVp : QKVh;
  const float* We  = chan ? We_p : We_h;
  float* wV = chan ? wV_p : wV_h;
  float* zz = chan ? z_p  : z_h;

  __shared__ __align__(16) float Es[64 * DIM];  // e tile (f32, async-staged)
  __shared__ __bf16 WtS[DIM * DIM];             // We^T
  __shared__ float  EeS[64 * DIM];              // projected edge feats
  __shared__ float  sS[64 * NHEAD];             // per-(edge,head) exp score
  __shared__ int    srcS[64], dstS[64];

  const int t  = threadIdx.x;
  const size_t e0 = (size_t)blockIdx.x * 64;

  // async stage of the HBM edge stream: 64x64 f32 = 16KB, 4 x b128 per thread
  const float* gE = e + e0 * DIM;
#pragma unroll
  for (int it = 0; it < 4; ++it) {
    int f4 = (t + it * 256) * 4;
    async_copy_b128(gE + f4, &Es[f4]);
  }
  for (int i = t; i < DIM * DIM; i += 256) {
    int k = i / DIM, n = i % DIM;
    WtS[n * DIM + k] = (__bf16)We[i];
  }
  if (t < 64) { srcS[t] = src[e0 + t]; dstS[t] = dst[e0 + t]; }
  async_wait0();
  __syncthreads();

  // WMMA: 4 row-tiles x 4 col-tiles of Ee; wave w -> row tile w>>1, col tiles {0,1}/{2,3}
  const int wave = t >> 5, lane = t & 31;
  const int et  = wave >> 1;
  const int ntb = (wave & 1) * 2;
  const float* Atile = Es + (et * 16) * DIM;
  v8f acc[2] = {};
#pragma unroll
  for (int kk = 0; kk < DIM; kk += 32) {
    v16bf a = load_a_frag_f32(Atile, DIM, kk, lane);
#pragma unroll
    for (int j = 0; j < 2; ++j) {
      v16bf b = load_b_frag(WtS + (ntb + j) * 16 * DIM, DIM, kk, lane);
      acc[j] = __builtin_amdgcn_wmma_f32_16x16x32_bf16(
          false, a, false, b, (short)0, acc[j], false, false);
    }
  }
  const int lh = lane >> 4, col = lane & 15;
#pragma unroll
  for (int j = 0; j < 2; ++j)
#pragma unroll
    for (int r = 0; r < 8; ++r)
      EeS[(et * 16 + lh * 8 + r) * DIM + (ntb + j) * 16 + col] = acc[j][r];
  __syncthreads();

  // scores: 64 edges x 8 heads = 512 tasks, 2 per thread
  const float inv_sqrt_hd = 0.35355339059327373f;  // 1/sqrt(8)
#pragma unroll
  for (int it = 0; it < 2; ++it) {
    int task = t + it * 256;
    int el = task >> 3, hd = task & 7;
    int s_ = srcS[el], d_ = dstS[el];
    float sum = 0.0f;
#pragma unroll
    for (int d = 0; d < HDIM; ++d) {
      int c = hd * HDIM + d;
      sum += QKV[(size_t)s_ * QS + 64 + c]   // K[src]
           * QKV[(size_t)d_ * QS + c]        // Q[dst]
           * EeS[el * DIM + c];
    }
    sum *= inv_sqrt_hd;
    sum = fminf(fmaxf(sum, -5.0f), 5.0f);
    float sv = expf(sum);
    sS[el * NHEAD + hd] = sv;
    atomicAdd(&zz[(size_t)d_ * NHEAD + hd], sv);
  }
  __syncthreads();

  // scatter V[src]*s into wV[dst]: 64 edges x 64 dims, 16 per thread
#pragma unroll
  for (int it = 0; it < 16; ++it) {
    int task = t + it * 256;
    int el = task >> 6, d = task & 63;
    int s_ = srcS[el], d_ = dstS[el];
    float val = QKV[(size_t)s_ * QS + 128 + d] * sS[el * NHEAD + (d >> 3)];
    atomicAdd(&wV[(size_t)d_ * DIM + d], val);
  }
}

// ---- BatchNorm (training-mode, biased stats) --------------------------------
__global__ void __launch_bounds__(256)
bn_stats_kernel(const float* __restrict__ X, int n,
                float* __restrict__ mean, float* __restrict__ var)
{
  __shared__ float ssum[256], ssq[256];
  int c = blockIdx.x;
  float s = 0.0f, q = 0.0f;
  for (int r = threadIdx.x; r < n; r += 256) {
    float v = X[(size_t)r * DIM + c];
    s += v; q += v * v;
  }
  ssum[threadIdx.x] = s; ssq[threadIdx.x] = q;
  __syncthreads();
  for (int w = 128; w > 0; w >>= 1) {
    if (threadIdx.x < w) {
      ssum[threadIdx.x] += ssum[threadIdx.x + w];
      ssq[threadIdx.x]  += ssq[threadIdx.x + w];
    }
    __syncthreads();
  }
  if (threadIdx.x == 0) {
    float m = ssum[0] / (float)n;
    mean[c] = m;
    var[c]  = ssq[0] / (float)n - m * m;
  }
}

__global__ void bn_apply_kernel(const float* __restrict__ X,
                                const float* __restrict__ mean, const float* __restrict__ var,
                                const float* __restrict__ g, const float* __restrict__ b,
                                float* __restrict__ Y, int n)
{
  int i = blockIdx.x * blockDim.x + threadIdx.x;
  if (i < n * DIM) {
    int c = i % DIM;
    Y[i] = (X[i] - mean[c]) * rsqrtf(var[c] + 1e-5f) * g[c] + b[c];
  }
}

// -----------------------------------------------------------------------------
extern "C" void kernel_launch(void* const* d_in, const int* in_sizes, int n_in,
                              void* d_out, int out_size, void* d_ws, size_t ws_size,
                              hipStream_t stream)
{
  (void)in_sizes; (void)n_in; (void)out_size; (void)ws_size;
  const float* h    = (const float*)d_in[0];
  const float* p    = (const float*)d_in[1];
  const float* e    = (const float*)d_in[2];
  // d_in[3] = snorm_n (unused by reference)
  const int*   src  = (const int*)d_in[4];
  const int*   dst  = (const int*)d_in[5];
  const float* Wq_h = (const float*)d_in[6];
  const float* Wk_h = (const float*)d_in[7];
  const float* We_h = (const float*)d_in[8];
  const float* Wv_h = (const float*)d_in[9];
  const float* Wq_p = (const float*)d_in[10];
  const float* Wk_p = (const float*)d_in[11];
  const float* We_p = (const float*)d_in[12];
  const float* Wv_p = (const float*)d_in[13];
  const float* Oh_W = (const float*)d_in[14];
  const float* Oh_b = (const float*)d_in[15];
  const float* Op_W = (const float*)d_in[16];
  const float* Op_b = (const float*)d_in[17];
  const float* bn1_g = (const float*)d_in[18];
  const float* bn1_b = (const float*)d_in[19];
  const float* bn2_g = (const float*)d_in[20];
  const float* bn2_b = (const float*)d_in[21];
  const float* W1   = (const float*)d_in[22];
  const float* b1   = (const float*)d_in[23];
  const float* W2   = (const float*)d_in[24];
  const float* b2   = (const float*)d_in[25];

  float* out_h = (float*)d_out;                         // hh: N x 64
  float* out_p = (float*)d_out + (size_t)NNODES * DIM;  // pp: N x 64

  // workspace layout (floats)
  const size_t ND = (size_t)NNODES * DIM;
  const size_t NH = (size_t)NNODES * NHEAD;
  float* ws = (float*)d_ws;
  size_t off = 0;
  float* hp_mid = ws + off; off += (size_t)NNODES * 128;  // hp, later FFN mid (aliased)
  float* QKVh = ws + off; off += (size_t)NNODES * QS;     // fused [Q|K|V], stride 192
  float* QKVp = ws + off; off += (size_t)NNODES * QS;
  float* wV_h = ws + off; off += ND;   // wV_h, wV_p, z_h, z_p contiguous for one memset
  float* wV_p = ws + off; off += ND;
  float* z_h  = ws + off; off += NH;
  float* z_p  = ws + off; off += NH;
  float* X1 = ws + off; off += ND;     // pre-BN residual buffer (reused)
  float* h2 = ws + off; off += ND;     // post-BN1
  float* Wc_h = ws + off; off += 128 * QS;  // packed QKV weights
  float* Wc_p = ws + off; off += 64 * QS;
  float* stats = ws + off; off += 256; // mean1,var1,mean2,var2

  const int EW_GRID = 12500;                        // N*DIM/256
  const int GEMM_GRID = (NNODES + 127) / 128;       // 391
  dim3 blk(256);

  // zero scatter accumulators (graph-capturable)
  (void)hipMemsetAsync(wV_h, 0, (2 * ND + 2 * NH) * sizeof(float), stream);

  // 1) hp = [h | p]; pack fused QKV weights
  concat_kernel<<<EW_GRID, blk, 0, stream>>>(h, p, hp_mid, NNODES);
  pack_qkv_kernel<<<(128 * QS + 255) / 256, blk, 0, stream>>>(Wq_h, Wk_h, Wv_h, Wc_h, 128);
  pack_qkv_kernel<<<(64 * QS + 255) / 256, blk, 0, stream>>>(Wq_p, Wk_p, Wv_p, Wc_p, 64);

  // 2) fused QKV projections (one WMMA GEMM per channel, A-tile staged once)
  gemm_kernel<128, QS, 0, false, false, false><<<GEMM_GRID, blk, 0, stream>>>(
      hp_mid, nullptr, Wc_h, nullptr, nullptr, QKVh, NNODES);
  gemm_kernel< 64, QS, 0, false, false, false><<<GEMM_GRID, blk, 0, stream>>>(
      p, nullptr, Wc_p, nullptr, nullptr, QKVp, NNODES);

  // 3) fused edge attention (async e-stage + Ee WMMA + score + exp + scatter)
  edge_attn_kernel<<<dim3(NEDGES / 64, 2), blk, 0, stream>>>(
      e, src, dst, QKVh, QKVp, We_h, We_p, wV_h, z_h, wV_p, z_p);

  // 4) output projections with fused wV/(z+eps) normalization:
  //    hh_pre = h + (att_h@Oh_W + Oh_b) ; pp = p + tanh(att_p@Op_W + Op_b)
  gemm_kernel<64, 64, 0, true, true, true><<<GEMM_GRID, blk, 0, stream>>>(
      wV_h, z_h, Oh_W, Oh_b, h, X1, NNODES);
  gemm_kernel<64, 64, 2, true, true, true><<<GEMM_GRID, blk, 0, stream>>>(
      wV_p, z_p, Op_W, Op_b, p, out_p, NNODES);

  // 5) BN1 -> h2
  bn_stats_kernel<<<DIM, blk, 0, stream>>>(X1, NNODES, stats, stats + 64);
  bn_apply_kernel<<<EW_GRID, blk, 0, stream>>>(X1, stats, stats + 64, bn1_g, bn1_b, h2, NNODES);

  // 6) FFN: mid = relu(h2@W1 + b1); X1 = h2 + mid@W2 + b2
  gemm_kernel< 64, 128, 1, true, false, false><<<GEMM_GRID, blk, 0, stream>>>(
      h2, nullptr, W1, b1, nullptr, hp_mid, NNODES);
  gemm_kernel<128,  64, 0, true, true, false><<<GEMM_GRID, blk, 0, stream>>>(
      hp_mid, nullptr, W2, b2, h2, X1, NNODES);

  // 7) BN2 -> out_h
  bn_stats_kernel<<<DIM, blk, 0, stream>>>(X1, NNODES, stats + 128, stats + 192);
  bn_apply_kernel<<<EW_GRID, blk, 0, stream>>>(X1, stats + 128, stats + 192, bn2_g, bn2_b, out_h, NNODES);
}
